// LSTMResidualLayer_28750511080159
// MI455X (gfx1250) — compile-verified, matrix operand order checked by reading, not run on hardware
//
#include <hip/hip_runtime.h>
#include <stdint.h>

// Problem constants (from reference): B=8, C=H=1024, T=2048, NC=64
#define BB 8
#define HH 1024
#define TT 2048
#define NC 64
#define G4H 4096
#define NBLK 64   // persistent LSTM workgroups (each owns 16 hidden cols)

typedef __attribute__((ext_vector_type(16))) _Float16 v16h;
typedef __attribute__((ext_vector_type(8)))  float    v8f;

union FragU { v16h v; uint32_t u[8]; };
union H2U   { uint32_t u; _Float16 h[2]; };

__device__ __forceinline__ float sigf(float x)  { return 1.0f / (1.0f + __expf(-x)); }
__device__ __forceinline__ float tanhf_(float x){ return 1.0f - 2.0f / (1.0f + __expf(2.0f * x)); }

// ---------------------------------------------------------------------------
// Zero init: h ping-pong buffers (16 rows x 1024 f16, rows 8..15 stay zero
// forever = padded M rows) + sync counter. Runs every call (graph replay safe).
// ---------------------------------------------------------------------------
__global__ void k_init(uint32_t* p, int n) {
  int i = blockIdx.x * blockDim.x + threadIdx.x;
  for (; i < n; i += gridDim.x * blockDim.x) p[i] = 0u;
}

// ---------------------------------------------------------------------------
// Prep: Wc = W_ih + W_hh folded to f16, pre-packed into per-lane WMMA
// B-fragment order.  Fragment id = ((g*8+w)*16 + G*4 + s); 256 u32 per frag
// stored as [d][lane].  Lane l holds column n = 16g + (l&15); k pattern per
// 16-bit operand layout: k = base + (d&3)*2 + 8*(l>>4) + 16*(d>>2), 2 halves.
// Also folds bias.
// ---------------------------------------------------------------------------
__global__ void k_prep(const float* __restrict__ Wih, const float* __restrict__ Whh,
                       const float* __restrict__ bih, const float* __restrict__ bhh,
                       uint32_t* __restrict__ wfrag, float* __restrict__ biasC) {
  int idx = blockIdx.x * blockDim.x + threadIdx.x;          // 8192 frags * 256
  if (idx < 8192 * 256) {
    int lane = idx & 31, d = (idx >> 5) & 7, frag = idx >> 8;
    int s = frag & 3, G = (frag >> 2) & 3, w = (frag >> 4) & 7, g = frag >> 7;
    int n = g * 16 + (lane & 15);
    int k = w * 128 + s * 32 + (d & 3) * 2 + 8 * (lane >> 4) + 16 * (d >> 2);
    int row = G * 1024 + n;
    H2U t;
    t.h[0] = (_Float16)(Wih[(size_t)row * HH + k]     + Whh[(size_t)row * HH + k]);
    t.h[1] = (_Float16)(Wih[(size_t)row * HH + k + 1] + Whh[(size_t)row * HH + k + 1]);
    wfrag[(size_t)frag * 256 + d * 32 + lane] = t.u;
  }
  if (idx < G4H) biasC[idx] = bih[idx] + bhh[idx];
}

// ---------------------------------------------------------------------------
// Persistent LSTM recurrence. 64 WGs x 256 thr (8 waves). WG g owns hidden
// cols [16g,16g+16). Wave w owns K slice [128w,128w+128). Weights live in
// VGPRs for the entire T=2048 loop; grid-wide sync via release/acquire atomic.
// All 4 A fragments are loaded up front each step so the compiler can clause
// the 8 global_load_b128 and stage s_wait_loadcnt against the 16 WMMAs.
// ---------------------------------------------------------------------------
__global__ void __launch_bounds__(256, 1)
k_lstm(const uint32_t* __restrict__ wfrag, const float* __restrict__ biasC,
       _Float16* __restrict__ hb0, _Float16* __restrict__ hb1,
       float* __restrict__ hs, uint32_t* __restrict__ syncc) {
  const int g = blockIdx.x, tid = threadIdx.x;
  const int w = tid >> 5, lane = tid & 31;
  const int laneM = lane & 15, khalf = lane >> 4;
  const int col = g * 16 + laneM;

  // Load all 16 B fragments (4 gates x 4 k-steps) into registers, once.
  FragU Bf[4][4];
#pragma unroll
  for (int G = 0; G < 4; ++G)
#pragma unroll
    for (int s = 0; s < 4; ++s) {
      const uint32_t* p = wfrag + ((size_t)((g * 8 + w) * 16 + G * 4 + s)) * 256 + lane;
#pragma unroll
      for (int d = 0; d < 8; ++d) Bf[G][s].u[d] = p[d * 32];
    }

  const float bi = biasC[col], bf = biasC[1024 + col],
              bg = biasC[2048 + col], bo = biasC[3072 + col];
  v8f cstate = {0.f,0.f,0.f,0.f,0.f,0.f,0.f,0.f};   // persistent cell state (wave 0)

  __shared__ float red[8][4][256];                  // [wave][gate][lane*8+r], 32 KB

  for (int t = 0; t < TT; ++t) {
    const _Float16* hin  = (t & 1) ? hb1 : hb0;
    _Float16*       hout = (t & 1) ? hb0 : hb1;

    // Fetch all 4 A fragments for this step (8x global_load_b128, one clause).
    FragU A[4];
#pragma unroll
    for (int s = 0; s < 4; ++s) {
      const int kbase = w * 128 + s * 32;
#pragma unroll
      for (int d = 0; d < 8; ++d) {
        int k = kbase + (d & 3) * 2 + 8 * khalf + 16 * (d >> 2);
        A[s].u[d] = *(const uint32_t*)(hin + laneM * HH + k);
      }
    }

    v8f acc[4];
#pragma unroll
    for (int G = 0; G < 4; ++G) acc[G] = (v8f){0.f,0.f,0.f,0.f,0.f,0.f,0.f,0.f};

#pragma unroll
    for (int s = 0; s < 4; ++s)
#pragma unroll
      for (int G = 0; G < 4; ++G)
        acc[G] = __builtin_amdgcn_wmma_f32_16x16x32_f16(
            false, A[s].v, false, Bf[G][s].v, (short)0, acc[G], false, false);

    // K-slice partials -> LDS
#pragma unroll
    for (int G = 0; G < 4; ++G)
#pragma unroll
      for (int r = 0; r < 8; ++r) red[w][G][lane * 8 + r] = acc[G][r];
    __syncthreads();

    if (w == 0) {
      float gi[8], gf[8], gg[8], go[8];
#pragma unroll
      for (int r = 0; r < 8; ++r) {
        gi[r] = red[0][0][lane * 8 + r];
        gf[r] = red[0][1][lane * 8 + r];
        gg[r] = red[0][2][lane * 8 + r];
        go[r] = red[0][3][lane * 8 + r];
      }
#pragma unroll
      for (int wv = 1; wv < 8; ++wv)
#pragma unroll
        for (int r = 0; r < 8; ++r) {
          gi[r] += red[wv][0][lane * 8 + r];
          gf[r] += red[wv][1][lane * 8 + r];
          gg[r] += red[wv][2][lane * 8 + r];
          go[r] += red[wv][3][lane * 8 + r];
        }
      float hv[8];
#pragma unroll
      for (int r = 0; r < 8; ++r) {
        float c2 = sigf(gf[r] + bf) * cstate[r] + sigf(gi[r] + bi) * tanhf_(gg[r] + bg);
        cstate[r] = c2;
        hv[r] = sigf(go[r] + bo) * tanhf_(c2);
      }
      if (lane < 16) {   // lanes 0..15 hold D rows M=0..7 (valid batch rows)
#pragma unroll
        for (int r = 0; r < 8; ++r) {
          hs[(size_t)t * (BB * HH) + r * HH + col] = hv[r];
          hout[r * HH + col] = (_Float16)hv[r];
        }
      }
    }

    // Grid-wide release/acquire sync between recurrence steps.
    if (tid == 0) {
      __hip_atomic_fetch_add(syncc, 1u, __ATOMIC_RELEASE, __HIP_MEMORY_SCOPE_AGENT);
      while (__hip_atomic_load(syncc, __ATOMIC_ACQUIRE, __HIP_MEMORY_SCOPE_AGENT)
             < (uint32_t)(NBLK * (t + 1)))
        __builtin_amdgcn_s_sleep(1);
    }
    __syncthreads();
  }
}

// ---------------------------------------------------------------------------
// frames_score = (hs @ W_fc.T + b_fc) * mask, via WMMA f16.
// hs viewed as [T*B=16384, 1024] row-major (row = t*8+b). One 16x16 tile per
// wave: 1024 row-tiles x 4 col-tiles = 4096 waves = 512 blocks.
// ---------------------------------------------------------------------------
__global__ void __launch_bounds__(256)
k_fc(const float* __restrict__ hs, const float* __restrict__ Wfc,
     const float* __restrict__ bfc, const float* __restrict__ masks,
     float* __restrict__ frames) {
  int wave = (blockIdx.x * blockDim.x + threadIdx.x) >> 5;
  int lane = threadIdx.x & 31;
  int rb = (wave >> 2) * 16, n0 = (wave & 3) * 16;
  int laneM = lane & 15, khalf = lane >> 4;

  v8f acc = {0.f,0.f,0.f,0.f,0.f,0.f,0.f,0.f};
  for (int kk = 0; kk < HH; kk += 32) {
    FragU Af, Bf;
#pragma unroll
    for (int d = 0; d < 8; ++d) {
      int k = kk + (d & 3) * 2 + 8 * khalf + 16 * (d >> 2);
      const float* ap = hs  + (size_t)(rb + laneM) * HH + k;
      const float* bp = Wfc + (size_t)(n0 + laneM) * HH + k;
      H2U a, b;
      a.h[0] = (_Float16)ap[0]; a.h[1] = (_Float16)ap[1];
      b.h[0] = (_Float16)bp[0]; b.h[1] = (_Float16)bp[1];
      Af.u[d] = a.u; Bf.u[d] = b.u;
    }
    acc = __builtin_amdgcn_wmma_f32_16x16x32_f16(
        false, Af.v, false, Bf.v, (short)0, acc, false, false);
  }
#pragma unroll
  for (int r = 0; r < 8; ++r) {
    int row = rb + r + 8 * khalf;          // 0..16383 (= t*8+b)
    int t = row >> 3, b = row & 7, n = n0 + laneM;
    frames[(size_t)b * (TT * NC) + (size_t)t * NC + n] =
        (acc[r] + bfc[n]) * masks[b * TT + t];
  }
}

// ---------------------------------------------------------------------------
// neck[b][c][t] = (x[b][c][t] + hs[t][b][c]) * mask[b][t]
// ---------------------------------------------------------------------------
__global__ void k_neck(const float* __restrict__ x, const float* __restrict__ hs,
                       const float* __restrict__ masks, float* __restrict__ neck) {
  int i = blockIdx.x * blockDim.x + threadIdx.x;
  const int N = BB * HH * TT;
  for (; i < N; i += gridDim.x * blockDim.x) {
    int b = i / (HH * TT), rem = i % (HH * TT);
    int c = rem / TT, t = rem % TT;
    neck[i] = (x[i] + hs[(size_t)t * (BB * HH) + b * HH + c]) * masks[b * TT + t];
  }
}

// ---------------------------------------------------------------------------
extern "C" void kernel_launch(void* const* d_in, const int* in_sizes, int n_in,
                              void* d_out, int out_size, void* d_ws, size_t ws_size,
                              hipStream_t stream) {
  const float* x     = (const float*)d_in[0];
  const float* masks = (const float*)d_in[1];
  const float* Wih   = (const float*)d_in[2];
  const float* Whh   = (const float*)d_in[3];
  const float* bih   = (const float*)d_in[4];
  const float* bhh   = (const float*)d_in[5];
  const float* Wfc   = (const float*)d_in[6];
  const float* bfc   = (const float*)d_in[7];

  char* ws = (char*)d_ws;
  uint32_t*  wfrag = (uint32_t*)(ws);                                  // 8 MB
  float*     biasC = (float*)   (ws + (8u << 20));                     // 16 KB
  _Float16*  hb0   = (_Float16*)(ws + (8u << 20) + (16u << 10));       // 32 KB
  _Float16*  hb1   = (_Float16*)(ws + (8u << 20) + (48u << 10));       // 32 KB
  uint32_t*  syncc = (uint32_t*)(ws + (8u << 20) + (80u << 10));       // 256 B
  float*     hs    = (float*)   (ws + (8u << 20) + (80u << 10) + 256); // 64 MB

  // Zero h buffers + sync counter (contiguous: 32K + 32K + 256 bytes)
  k_init<<<65, 256, 0, stream>>>((uint32_t*)hb0, (32768 + 32768 + 256) / 4);
  // Fold & pack weights/bias
  k_prep<<<8192, 256, 0, stream>>>(Wih, Whh, bih, bhh, wfrag, biasC);
  // Persistent recurrence
  k_lstm<<<NBLK, 256, 0, stream>>>(wfrag, biasC, hb0, hb1, hs, syncc);

  float* neck   = (float*)d_out;
  float* frames = neck + (size_t)BB * HH * TT;
  k_fc  <<<512,  256, 0, stream>>>(hs, Wfc, bfc, masks, frames);
  k_neck<<<8192, 256, 0, stream>>>(x, hs, masks, neck);
}
